// MixtureOfExperts_29171417875244
// MI455X (gfx1250) — matrix operand order repaired and checked by reading, not compiled, not run on hardware
//
#include <hip/hip_runtime.h>
#include <hip/hip_bf16.h>

typedef unsigned int  u32;
typedef unsigned short u16;

typedef __attribute__((ext_vector_type(16))) __bf16 v16bf;
typedef __attribute__((ext_vector_type(8)))  float  v8f;
typedef __attribute__((ext_vector_type(4)))  int    v4i;

#define AS_GLOBAL __attribute__((address_space(1)))
#define AS_LDS    __attribute__((address_space(3)))

// ---------------- problem constants ----------------
constexpr int TOK = 4096;   // B*S
constexpr int DM  = 512;    // d_model
constexpr int DF  = 2048;   // d_ff
constexpr int NE  = 8;      // experts
constexpr int RMAX = 9216;  // 8192 assignments + 8*128 pad, = 72*128

// ---------------- workspace layout (bytes) ----------------
constexpr size_t WS_CTR      = 0;                                   // cnt[8], fill[8], base[9]
constexpr size_t WS_TOPK_IDX = 256;
constexpr size_t WS_TOPK_W   = WS_TOPK_IDX + (size_t)TOK * 2 * 4;
constexpr size_t WS_ROWOF    = WS_TOPK_W   + (size_t)TOK * 2 * 4;
constexpr size_t WS_TOKROW   = WS_ROWOF    + (size_t)TOK * 2 * 4;
constexpr size_t WS_XG       = (WS_TOKROW + (size_t)RMAX * 4 + 255) & ~(size_t)255;
constexpr size_t WS_W1T      = WS_XG  + (size_t)RMAX * DM * 2;
constexpr size_t WS_W2T      = WS_W1T + (size_t)NE * DF * DM * 2;
constexpr size_t WS_H        = WS_W2T + (size_t)NE * DM * DF * 2;
constexpr size_t WS_Y        = WS_H   + (size_t)RMAX * DF * 2;

#if __has_builtin(__builtin_amdgcn_global_load_async_to_lds_b128)
#define MOE_ASYNC_LDS 1
#else
#define MOE_ASYNC_LDS 0
#endif

__device__ __forceinline__ void wait_async0() {
#if MOE_ASYNC_LDS
#if __has_builtin(__builtin_amdgcn_s_wait_asynccnt)
  __builtin_amdgcn_s_wait_asynccnt(0);
#else
  asm volatile("s_wait_asynccnt 0" ::: "memory");
#endif
#endif
}

// copy 16 bytes global -> LDS (async-to-LDS on gfx1250 when available)
__device__ __forceinline__ void copy16(const u16* __restrict__ g, u16* l) {
#if MOE_ASYNC_LDS
  __builtin_amdgcn_global_load_async_to_lds_b128(
      (AS_GLOBAL v4i*)g, (AS_LDS v4i*)l, 0, 0);
#else
  *(uint4*)l = *(const uint4*)g;
#endif
}

__device__ __forceinline__ u16 bf16_rne(float f) {
  u32 u = __float_as_uint(f);
  u += 0x7FFFu + ((u >> 16) & 1u);
  return (u16)(u >> 16);
}

// ---------------- 1) gating: one wave per token ----------------
__global__ __launch_bounds__(256) void gate_kernel(
    const float* __restrict__ x, const float* __restrict__ Wg,
    const float* __restrict__ bg, int* __restrict__ topk_idx,
    float* __restrict__ topk_w) {
  int t    = blockIdx.x * 8 + (threadIdx.x >> 5);
  int lane = threadIdx.x & 31;
  if (t >= TOK) return;
  const float* xr = x + (size_t)t * DM;
  float acc[NE];
#pragma unroll
  for (int e = 0; e < NE; ++e) acc[e] = 0.f;
  for (int i = lane; i < DM; i += 32) {
    float xv = xr[i];
    const float* wr = Wg + (size_t)i * NE;
#pragma unroll
    for (int e = 0; e < NE; ++e) acc[e] += xv * wr[e];
  }
#pragma unroll
  for (int e = 0; e < NE; ++e)
    for (int off = 16; off; off >>= 1) acc[e] += __shfl_xor(acc[e], off, 32);
  if (lane == 0) {
    float l[NE];
#pragma unroll
    for (int e = 0; e < NE; ++e) l[e] = acc[e] + bg[e];
    int i0 = 0;
#pragma unroll
    for (int e = 1; e < NE; ++e) if (l[e] > l[i0]) i0 = e;
    int i1 = (i0 == 0) ? 1 : 0;
#pragma unroll
    for (int e = 0; e < NE; ++e) if (e != i0 && l[e] > l[i1]) i1 = e;
    // softmax-top2 renormalized == softmax over the two winning logits
    float ex = __expf(l[i1] - l[i0]);
    float w0 = 1.f / (1.f + ex);
    topk_idx[t * 2 + 0] = i0;
    topk_idx[t * 2 + 1] = i1;
    topk_w[t * 2 + 0] = w0;
    topk_w[t * 2 + 1] = 1.f - w0;
  }
}

// ---------------- 2) routing ----------------
__global__ void count_kernel(const int* __restrict__ topk_idx, u32* __restrict__ cnt) {
  int t = blockIdx.x * 256 + threadIdx.x;
  if (t < TOK) {
    atomicAdd(&cnt[topk_idx[t * 2 + 0]], 1u);
    atomicAdd(&cnt[topk_idx[t * 2 + 1]], 1u);
  }
}

__global__ void scan_kernel(const u32* __restrict__ cnt, u32* __restrict__ base) {
  if (threadIdx.x == 0 && blockIdx.x == 0) {
    u32 s = 0;
    for (int e = 0; e < NE; ++e) { base[e] = s; s += (cnt[e] + 127u) & ~127u; }
    base[NE] = s;  // totalRows (128-aligned per expert)
  }
}

__global__ void assign_kernel(const int* __restrict__ topk_idx,
                              const u32* __restrict__ base, u32* __restrict__ fill,
                              int* __restrict__ rowOf, int* __restrict__ tokenOfRow) {
  int t = blockIdx.x * 256 + threadIdx.x;
  if (t < TOK) {
#pragma unroll
    for (int k = 0; k < 2; ++k) {
      int e = topk_idx[t * 2 + k];
      u32 s = atomicAdd(&fill[e], 1u);
      int r = (int)(base[e] + s);
      rowOf[t * 2 + k] = r;
      tokenOfRow[r] = t;
    }
  }
}

// ---------------- 3) weight transpose + fp32->bf16 ----------------
// src: [batch][rows][cols] fp32 -> dst: [batch][cols][rows] bf16
__global__ __launch_bounds__(256) void transpose_bf16_kernel(
    const float* __restrict__ src, u16* __restrict__ dst, int rows, int cols) {
  __shared__ float tile[32][33];
  size_t mat = (size_t)rows * cols;
  src += (size_t)blockIdx.z * mat;
  dst += (size_t)blockIdx.z * mat;
  int c0 = blockIdx.x * 32, r0 = blockIdx.y * 32;
  int tx = threadIdx.x, ty = threadIdx.y;  // 32 x 8
#pragma unroll
  for (int i = 0; i < 4; ++i)
    tile[ty + 8 * i][tx] = src[(size_t)(r0 + ty + 8 * i) * cols + c0 + tx];
  __syncthreads();
#pragma unroll
  for (int i = 0; i < 4; ++i)
    dst[(size_t)(c0 + ty + 8 * i) * rows + r0 + tx] = bf16_rne(tile[tx][ty + 8 * i]);
}

// ---------------- 4) gather tokens -> compacted bf16 rows ----------------
__global__ __launch_bounds__(256) void gather_kernel(
    const float* __restrict__ x, const u32* __restrict__ base,
    const u32* __restrict__ cnt, const int* __restrict__ tokenOfRow,
    u16* __restrict__ xg) {
  int r = blockIdx.x;
  if ((u32)r >= base[NE]) return;
  int e = 0;
  while (base[e + 1] <= (u32)r) ++e;
  u32 i = (u32)r - base[e];
  if (i < cnt[e]) {
    const float* xr = x + (size_t)tokenOfRow[r] * DM;
    for (int d = threadIdx.x; d < DM; d += 256)
      xg[(size_t)r * DM + d] = bf16_rne(xr[d]);
  } else {
    for (int d = threadIdx.x; d < DM; d += 256)
      xg[(size_t)r * DM + d] = 0;
  }
}

// ---------------- 5) grouped GEMM with bf16 WMMA ----------------
// A: [RMAX][KD] bf16 (rows grouped by expert, 128-aligned), B: [NE][ND][KD] bf16
// C = A*B^T + bias. Block tile 128x128, 8 waves, wave tile 32x64.
// Double-buffered LDS; async global->LDS copies (ASYNCcnt) when available.
template <int KD, int ND, bool DO_GELU, bool OUT_BF16>
__global__ __launch_bounds__(256) void moe_gemm_kernel(
    const u16* __restrict__ Abuf, const u16* __restrict__ Bbuf,
    const float* __restrict__ bias, void* __restrict__ Obuf,
    const u32* __restrict__ base) {
  constexpr int LDA = 40;          // padded LDS row stride (u16): 80B, bank-friendly
  constexpr int BUF = 128 * LDA;   // one A (or B) tile buffer, in u16
  __shared__ __attribute__((aligned(16))) u16 lds[4 * BUF];  // {buf0,buf1} x {A,B}

  const int r0 = blockIdx.y * 128;
  if ((u32)r0 >= base[NE]) return;  // uniform early-exit (EXEC stays all-ones)
  int e = 0;
  while (base[e + 1] <= (u32)r0) ++e;  // 128-pad => tile is expert-uniform

  const int n0 = blockIdx.x * 128;
  const u16* Ap = Abuf + (size_t)r0 * KD;
  const u16* Bp = Bbuf + ((size_t)e * ND + n0) * KD;

  const int tid  = threadIdx.x;
  const int lane = tid & 31;
  const int wave = tid >> 5;
  const int wm   = (wave & 3) * 32;   // 4 wave-rows  x 32
  const int wn   = (wave >> 2) * 64;  // 2 wave-cols  x 64
  const int l15  = lane & 15;
  const int half = lane >> 4;

  v8f acc[2][4];
#pragma unroll
  for (int mi = 0; mi < 2; ++mi)
#pragma unroll
    for (int ni = 0; ni < 4; ++ni)
      acc[mi][ni] = (v8f){0.f, 0.f, 0.f, 0.f, 0.f, 0.f, 0.f, 0.f};

  // loader: 128 rows x 64B per tile; 256 threads x 2 slots (A) + 2 slots (B)
  const int rowa = tid >> 2;       // 0..63
  const int ko   = (tid & 3) * 8;  // 16B chunk within 32-elem row

  auto issue = [&](int kc, int bufsel) {
    u16* Ad = &lds[bufsel * 2 * BUF];
    u16* Bd = &lds[bufsel * 2 * BUF + BUF];
    copy16(&Ap[(size_t)rowa * KD + kc + ko],        &Ad[rowa * LDA + ko]);
    copy16(&Ap[(size_t)(rowa + 64) * KD + kc + ko], &Ad[(rowa + 64) * LDA + ko]);
    copy16(&Bp[(size_t)rowa * KD + kc + ko],        &Bd[rowa * LDA + ko]);
    copy16(&Bp[(size_t)(rowa + 64) * KD + kc + ko], &Bd[(rowa + 64) * LDA + ko]);
  };

  issue(0, 0);
  wait_async0();
  __syncthreads();

  for (int kc = 0; kc < KD; kc += 32) {
    const int cur = (kc >> 5) & 1;
    if (kc + 32 < KD) issue(kc + 32, cur ^ 1);  // prefetch next chunk

    const u16* Ac = &lds[cur * 2 * BUF];
    const u16* Bc = &lds[cur * 2 * BUF + BUF];

    v16bf afrag[2], bfrag[4];
#pragma unroll
    for (int mi = 0; mi < 2; ++mi) {  // A lane layout: K = 8*half+{0..7}, 16+8*half+{0..7}
      const int m = wm + mi * 16 + l15;
      ((uint4*)&afrag[mi])[0] = *(const uint4*)&Ac[m * LDA + half * 8];
      ((uint4*)&afrag[mi])[1] = *(const uint4*)&Ac[m * LDA + 16 + half * 8];
    }
#pragma unroll
    for (int ni = 0; ni < 4; ++ni) {  // B lane layout: K = 8*half+{0..7}, 16+8*half+{0..7}
      const int n = wn + ni * 16 + l15;
      ((uint4*)&bfrag[ni])[0] = *(const uint4*)&Bc[n * LDA + half * 8];
      ((uint4*)&bfrag[ni])[1] = *(const uint4*)&Bc[n * LDA + 16 + half * 8];
    }
#pragma unroll
    for (int mi = 0; mi < 2; ++mi)
#pragma unroll
      for (int ni = 0; ni < 4; ++ni)
        acc[mi][ni] = __builtin_amdgcn_wmma_f32_16x16x32_bf16(
            false, afrag[mi], false, bfrag[ni], (short)0, acc[mi][ni], false, false);

    wait_async0();     // prefetch for next chunk complete
    __syncthreads();   // everyone done reading cur; next iter may overwrite cur
  }

  // epilogue: C layout M = v + 8*half, N = lane&15
#pragma unroll
  for (int mi = 0; mi < 2; ++mi) {
#pragma unroll
    for (int ni = 0; ni < 4; ++ni) {
      const int gn = n0 + wn + ni * 16 + l15;
      const float bv = bias[(size_t)e * ND + gn];
#pragma unroll
      for (int v = 0; v < 8; ++v) {
        const int gm = r0 + wm + mi * 16 + half * 8 + v;
        float val = acc[mi][ni][v] + bv;
        if (DO_GELU) val = 0.5f * val * (1.0f + erff(val * 0.70710678118654752f));
        if (OUT_BF16)
          ((u16*)Obuf)[(size_t)gm * ND + gn] = bf16_rne(val);
        else
          ((float*)Obuf)[(size_t)gm * ND + gn] = val;
      }
    }
  }
}

// ---------------- 6) combine ----------------
__global__ __launch_bounds__(256) void combine_kernel(
    const float* __restrict__ Y, const int* __restrict__ rowOf,
    const float* __restrict__ topk_w, float* __restrict__ out) {
  int idx = blockIdx.x * 256 + threadIdx.x;  // over TOK*DM
  int t = idx >> 9;                          // DM == 512
  int d = idx & 511;
  int r0 = rowOf[t * 2 + 0], r1 = rowOf[t * 2 + 1];
  out[idx] = topk_w[t * 2 + 0] * Y[(size_t)r0 * DM + d] +
             topk_w[t * 2 + 1] * Y[(size_t)r1 * DM + d];
}

// ---------------- host ----------------
extern "C" void kernel_launch(void* const* d_in, const int* in_sizes, int n_in,
                              void* d_out, int out_size, void* d_ws, size_t ws_size,
                              hipStream_t stream) {
  (void)in_sizes; (void)n_in; (void)out_size; (void)ws_size;
  const float* x  = (const float*)d_in[0];
  const float* Wg = (const float*)d_in[1];
  const float* bg = (const float*)d_in[2];
  const float* W1 = (const float*)d_in[3];
  const float* b1 = (const float*)d_in[4];
  const float* W2 = (const float*)d_in[5];
  const float* b2 = (const float*)d_in[6];
  float* out = (float*)d_out;

  char* ws = (char*)d_ws;
  u32* cnt        = (u32*)(ws + WS_CTR);
  u32* fill       = cnt + 8;
  u32* base       = fill + 8;
  int* topk_idx   = (int*)(ws + WS_TOPK_IDX);
  float* topk_w   = (float*)(ws + WS_TOPK_W);
  int* rowOf      = (int*)(ws + WS_ROWOF);
  int* tokenOfRow = (int*)(ws + WS_TOKROW);
  u16* xg         = (u16*)(ws + WS_XG);
  u16* W1T        = (u16*)(ws + WS_W1T);
  u16* W2T        = (u16*)(ws + WS_W2T);
  u16* H          = (u16*)(ws + WS_H);
  float* Y        = (float*)(ws + WS_Y);

  (void)hipMemsetAsync(ws, 0, 256, stream);  // cnt/fill/base

  gate_kernel<<<TOK / 8, 256, 0, stream>>>(x, Wg, bg, topk_idx, topk_w);
  count_kernel<<<TOK / 256, 256, 0, stream>>>(topk_idx, cnt);
  scan_kernel<<<1, 64, 0, stream>>>(cnt, base);
  assign_kernel<<<TOK / 256, 256, 0, stream>>>(topk_idx, base, fill, rowOf, tokenOfRow);

  // W1 [E][DM][DF] -> W1T [E][DF][DM] ; W2 [E][DF][DM] -> W2T [E][DM][DF]
  transpose_bf16_kernel<<<dim3(DF / 32, DM / 32, NE), dim3(32, 8), 0, stream>>>(W1, W1T, DM, DF);
  transpose_bf16_kernel<<<dim3(DM / 32, DF / 32, NE), dim3(32, 8), 0, stream>>>(W2, W2T, DF, DM);

  gather_kernel<<<RMAX, 256, 0, stream>>>(x, base, cnt, tokenOfRow, xg);

  moe_gemm_kernel<DM, DF, true, true>
      <<<dim3(DF / 128, RMAX / 128), 256, 0, stream>>>(xg, W1T, b1, (void*)H, base);
  moe_gemm_kernel<DF, DM, false, false>
      <<<dim3(DM / 128, RMAX / 128), 256, 0, stream>>>(H, W2T, b2, (void*)Y, base);

  combine_kernel<<<(TOK * DM) / 256, 256, 0, stream>>>(Y, rowOf, topk_w, out);
}
